// EncoderALSTM_68186900791776
// MI455X (gfx1250) — matrix operand order repaired and checked by reading, not compiled, not run on hardware
//
#include <hip/hip_runtime.h>
#include <hip/hip_bf16.h>

// ---------------------------------------------------------------------------
// Problem constants (reference: T,B,H,I = 128,128,256,256)
// ---------------------------------------------------------------------------
#define TT   128
#define BB   128
#define HH   256
#define II   256
#define GG   1024          // 4*H
#define BT   16            // batch tile per workgroup
#define NTHR 512

typedef _Float16     v16h  __attribute__((ext_vector_type(16)));
typedef float        v8f   __attribute__((ext_vector_type(8)));
typedef unsigned int u32x4 __attribute__((ext_vector_type(4)));
typedef float        f32x4 __attribute__((ext_vector_type(4)));

union Frag {
    v16h     v;
    _Float16 h[16];
    u32x4    q[2];
};
union H8 {
    u32x4    q;
    _Float16 h[8];
};

// D = A(16x32 f16) x B(32x16 f16) + C(16x16 f32)
static __device__ __forceinline__ v8f wmma_f16(v16h a, v16h b, v8f c) {
    return __builtin_amdgcn_wmma_f32_16x16x32_f16(false, a, false, b, (short)0, c, false, false);
}

// A-matrix fragment, 16x32, from a row-major f16 array with leading dim `ld`.
// Lane L (L<16): row=L, elems[0..7]=K+0..7, elems[8..15]=K+16..23
// Lane L>=16  : row=L-16, elems[0..7]=K+8..15, elems[8..15]=K+24..31
static __device__ __forceinline__ v16h load_a_frag(const _Float16* A, int ld,
                                                   int kofs, int lane) {
    int row = lane & 15;
    int hb  = (lane >> 4) << 3;             // 0 or 8
    const _Float16* p = A + (size_t)row * ld + kofs + hb;
    Frag f;
    f.q[0] = *(const u32x4*)(p);            // K = kofs+hb    .. +7  (16B aligned)
    f.q[1] = *(const u32x4*)(p + 16);       // K = kofs+16+hb .. +7
    return f.v;
}

// Same layout but sourced from f32 memory (converted on the fly).
static __device__ __forceinline__ v16h load_a_frag_f32(const float* A, int ld,
                                                       int row0, int kofs, int lane) {
    int row = row0 + (lane & 15);
    int hb  = (lane >> 4) << 3;
    const float* p = A + (size_t)row * ld + kofs + hb;
    Frag f;
#pragma unroll
    for (int j = 0; j < 8; ++j) f.h[j]     = (_Float16)p[j];
#pragma unroll
    for (int j = 0; j < 8; ++j) f.h[8 + j] = (_Float16)p[16 + j];
    return f.v;
}

// B-matrix fragment, 32(K)x16(N), where B[k][n] = W[nbase+n][k] and W is a
// row-major [*, ldk] f16 weight matrix (i.e. the GEMM computes X @ W^T).
// Lane L: col = L&15, elems[0..7] = K = (L>=16?8:0)+0..7, elems[8..15] = K+16.. (same as A).
static __device__ __forceinline__ v16h load_b_frag(const _Float16* W, int ldk,
                                                   int nbase, int kofs, int lane) {
    int col = lane & 15;
    int kb  = (lane >> 4) << 3;             // 0 or 8
    const _Float16* p = W + (size_t)(nbase + col) * ldk + kofs + kb;
    Frag f;
    f.q[0] = *(const u32x4*)(p);
    f.q[1] = *(const u32x4*)(p + 16);
    return f.v;
}

static __device__ __forceinline__ float fast_tanh(float x) {
    // 1 - 2/(e^{2x}+1): saturates cleanly to +/-1 at both ends.
    return 1.f - 2.f / (__expf(2.f * x) + 1.f);
}
static __device__ __forceinline__ float fast_sigmoid(float x) {
    return 1.f / (1.f + __expf(-x));
}

// ---------------------------------------------------------------------------
// Kernel 1: f32 -> f16 weight conversion
// ---------------------------------------------------------------------------
__global__ void f32_to_f16_kernel(const float* __restrict__ src,
                                  _Float16* __restrict__ dst, int n) {
    int i = blockIdx.x * 256 + threadIdx.x;
    if (i < n) dst[i] = (_Float16)src[i];
}

// ---------------------------------------------------------------------------
// Kernel 2: pre_ih = embs @ Wih^T + bih + bhh, stored in tile-major
// D-fragment layout: pre[((mt*64+nt)*32 + lane)*8 + r] so both this kernel's
// stores and the recurrence's accumulator-init loads are contiguous 32B/lane.
// ---------------------------------------------------------------------------
__global__ __launch_bounds__(128) void pre_gates_kernel(
        const float* __restrict__ embs,       // [T*B, I] f32
        const _Float16* __restrict__ WihH,    // [4H, I] f16
        const float* __restrict__ bih,
        const float* __restrict__ bhh,
        float* __restrict__ pre) {            // tile-major, see above
    const int lane = threadIdx.x & 31;
    const int wv   = threadIdx.x >> 5;
    const int tile = blockIdx.x * 4 + wv;     // 65536 tiles = (mt<<6)|nt
    const int nt   = tile & 63;               // 64 N tiles (G=1024)
    const int mt   = tile >> 6;               // 1024 M tiles (T*B=16384)

    v8f acc = {};
#pragma unroll
    for (int kt = 0; kt < II / 32; ++kt) {
        v16h a = load_a_frag_f32(embs, II, mt * 16, kt * 32, lane);
        v16h b = load_b_frag(WihH, II, nt * 16, kt * 32, lane);
        acc = wmma_f16(a, b, acc);
    }
    const int n = lane & 15;
    const int g = nt * 16 + n;
    const float bsum = bih[g] + bhh[g];
    float* dst = pre + ((size_t)tile * 32 + lane) * 8;
#pragma unroll
    for (int r = 0; r < 8; ++r) dst[r] = acc[r] + bsum;
}

// ---------------------------------------------------------------------------
// Kernel 3: the recurrence. 8 blocks x 512 threads; block j owns batch rows
// [16j, 16j+16). No inter-block communication. Wave w owns h-columns
// [16w, 16w+16): its i/f/g/o WMMA accumulators + persistent c registers.
// Attention history is kept both in f32 (output hs) and a f16 mirror (hidsH)
// that serves the score/ctx streams at half the L2 traffic.
// ---------------------------------------------------------------------------
__global__ __launch_bounds__(NTHR) void alstm_kernel(
        const int* __restrict__ lens,
        const _Float16* __restrict__ WhhH,    // [4H, H] f16
        const _Float16* __restrict__ WaH,     // [H, 2H] f16
        const float* __restrict__ ba,         // [H]
        const float* __restrict__ pre,        // tile-major pre_ih
        _Float16* __restrict__ hidsH,         // [T, B, H] f16 history mirror
        float* __restrict__ hs,               // [T, B, H] (output 0)
        float* __restrict__ hfin,             // [B, H]
        float* __restrict__ cfin) {           // [B, H]
    __shared__ _Float16  sh_hbf[BT][HH];      // h / h_in f16 (WMMA A for Whh)
    __shared__ _Float16  sh_cat[BT][2 * HH];  // [h | ctx] f16 (WMMA A for Wa)
    __shared__ float     sh_sc[TT][BT];       // scores / softmax weights
    __shared__ int       sh_len[BT];

    const int tid  = threadIdx.x;
    const int lane = tid & 31;
    const int wv   = tid >> 5;                // 0..15
    const int bb   = blockIdx.x * BT;

    // init: h = c = 0
    for (int i = tid; i < BT * HH; i += NTHR)
        ((_Float16*)sh_hbf)[i] = (_Float16)0.f;
    if (tid < BT) sh_len[tid] = lens[bb + tid];
    v8f creg = {};
    __syncthreads();

    for (int t = 0; t < TT; ++t) {
        // ---------------- attention (t >= 1) ----------------
        if (t > 0) {
            // scores[tp][b] = h[b] . hids[tp][bb+b]   (f16 x f16 -> f32)
            const int npairs = t * BT;
            for (int idx = tid; idx < npairs; idx += NTHR) {
                int tp = idx >> 4, b = idx & 15;
                const u32x4* hv = (const u32x4*)&sh_hbf[b][0];
                const u32x4* gv = (const u32x4*)(hidsH + ((size_t)tp * BB + bb + b) * HH);
                float a0 = 0.f, a1 = 0.f, a2 = 0.f, a3 = 0.f;
                for (int k = 0; k < HH / 8; ++k) {
                    H8 x, y;
                    x.q = hv[k];
                    y.q = gv[k];
                    a0 += (float)x.h[0] * (float)y.h[0] + (float)x.h[1] * (float)y.h[1];
                    a1 += (float)x.h[2] * (float)y.h[2] + (float)x.h[3] * (float)y.h[3];
                    a2 += (float)x.h[4] * (float)y.h[4] + (float)x.h[5] * (float)y.h[5];
                    a3 += (float)x.h[6] * (float)y.h[6] + (float)x.h[7] * (float)y.h[7];
                }
                sh_sc[tp][b] = (a0 + a1) + (a2 + a3);
            }
            __syncthreads();

            // softmax over tp in [0,t), per batch column
            if (tid < BT) {
                float mx = -3.0e38f;
                for (int tp = 0; tp < t; ++tp) mx = fmaxf(mx, sh_sc[tp][tid]);
                float s = 0.f;
                for (int tp = 0; tp < t; ++tp) {
                    float e = __expf(sh_sc[tp][tid] - mx);
                    sh_sc[tp][tid] = e;
                    s += e;
                }
                float inv = 1.f / s;
                for (int tp = 0; tp < t; ++tp) sh_sc[tp][tid] *= inv;
            }
            __syncthreads();

            // ctx[b][h] = sum_tp w[tp][b] * hids[tp][bb+b][h]; build [h|ctx] f16
            {
                int b  = tid >> 5;            // wave -> batch row
                int h0 = (tid & 31) * 8;
                float cacc[8] = {0.f, 0.f, 0.f, 0.f, 0.f, 0.f, 0.f, 0.f};
                for (int tp = 0; tp < t; ++tp) {
                    float wgt = sh_sc[tp][b];
                    H8 y;
                    y.q = *(const u32x4*)(hidsH + ((size_t)tp * BB + bb + b) * HH + h0);
#pragma unroll
                    for (int j = 0; j < 8; ++j) cacc[j] += wgt * (float)y.h[j];
                }
#pragma unroll
                for (int j = 0; j < 8; ++j) {
                    sh_cat[b][h0 + j]      = sh_hbf[b][h0 + j];
                    sh_cat[b][HH + h0 + j] = (_Float16)cacc[j];
                }
            }
            __syncthreads();

            // h_att = tanh([h|ctx] @ Wa^T + ba); wave w -> n-tile w (16 cols)
            {
                v8f acc = {};
#pragma unroll
                for (int kt = 0; kt < (2 * HH) / 32; ++kt) {
                    v16h a = load_a_frag(&sh_cat[0][0], 2 * HH, kt * 32, lane);
                    v16h b = load_b_frag(WaH, 2 * HH, wv * 16, kt * 32, lane);
                    acc = wmma_f16(a, b, acc);
                }
                const int n  = lane & 15;
                const int hi = lane >> 4;
                const float ban = ba[wv * 16 + n];
#pragma unroll
                for (int r = 0; r < 8; ++r) {
                    int m = r + 8 * hi;
                    sh_hbf[m][wv * 16 + n] = (_Float16)fast_tanh(acc[r] + ban);
                }
            }
            __syncthreads();
        }
        // (t == 0: sh_hbf is still the zero h -> h_in = h = 0, matching ref)

        // ---------------- gates = pre_ih[t] + h_in @ Whh^T ----------------
        v8f ai, af_, ag, ao;
        {
            // tile-major pre: contiguous 32B per lane per gate tile
            const size_t mt  = (size_t)t * (BB / 16) + blockIdx.x;
            const float* p0 = pre + (((mt << 6) |      wv) * 32 + lane) * 8;
            const float* p1 = pre + (((mt << 6) | (16 + wv)) * 32 + lane) * 8;
            const float* p2 = pre + (((mt << 6) | (32 + wv)) * 32 + lane) * 8;
            const float* p3 = pre + (((mt << 6) | (48 + wv)) * 32 + lane) * 8;
#pragma unroll
            for (int r = 0; r < 8; ++r) {
                ai[r]  = p0[r];
                af_[r] = p1[r];
                ag[r]  = p2[r];
                ao[r]  = p3[r];
            }
        }
#pragma unroll
        for (int kt = 0; kt < HH / 32; ++kt) {
            v16h a = load_a_frag(&sh_hbf[0][0], HH, kt * 32, lane);
            ai  = wmma_f16(a, load_b_frag(WhhH, HH,          wv * 16, kt * 32, lane), ai);
            af_ = wmma_f16(a, load_b_frag(WhhH, HH, HH     + wv * 16, kt * 32, lane), af_);
            ag  = wmma_f16(a, load_b_frag(WhhH, HH, 2 * HH + wv * 16, kt * 32, lane), ag);
            ao  = wmma_f16(a, load_b_frag(WhhH, HH, 3 * HH + wv * 16, kt * 32, lane), ao);
        }
        __syncthreads();   // all waves done reading sh_hbf before overwrite

        // ---------------- LSTM cell elementwise, in registers -------------
        {
            const int n  = lane & 15;
            const int hi = lane >> 4;
            const int hc = wv * 16 + n;
#pragma unroll
            for (int r = 0; r < 8; ++r) {
                int m = r + 8 * hi;
                float iv = fast_sigmoid(ai[r]);
                float fv = fast_sigmoid(af_[r]);
                float gv = fast_tanh(ag[r]);
                float ov = fast_sigmoid(ao[r]);
                float cn = fv * creg[r] + iv * gv;
                float hn = ov * fast_tanh(cn);
                creg[r] = cn;
                sh_hbf[m][hc] = (_Float16)hn;
                size_t gidx = ((size_t)t * BB + bb + m) * HH + hc;
                hs[gidx]    = hn;
                hidsH[gidx] = (_Float16)hn;
                if (t == sh_len[m] - 1) {
                    hfin[(size_t)(bb + m) * HH + hc] = hn;
                    cfin[(size_t)(bb + m) * HH + hc] = cn;
                }
            }
        }
        __threadfence();   // make hs[t]/hidsH[t] visible to later global loads
        __syncthreads();
    }
}

// ---------------------------------------------------------------------------
// Host-side launcher
// ---------------------------------------------------------------------------
extern "C" void kernel_launch(void* const* d_in, const int* in_sizes, int n_in,
                              void* d_out, int out_size, void* d_ws, size_t ws_size,
                              hipStream_t stream) {
    (void)in_sizes; (void)n_in; (void)out_size; (void)ws_size;
    const float* embs = (const float*)d_in[0];   // [T,B,I]
    const int*   lens = (const int*)d_in[1];     // [B]
    const float* Wih  = (const float*)d_in[2];   // [4H,I]
    const float* Whh  = (const float*)d_in[3];   // [4H,H]
    const float* bih  = (const float*)d_in[4];   // [4H]
    const float* bhh  = (const float*)d_in[5];   // [4H]
    const float* Wa   = (const float*)d_in[6];   // [H,2H]
    const float* ba   = (const float*)d_in[7];   // [H]

    float* out  = (float*)d_out;
    float* hs   = out;                                   // [T,B,H]
    float* hfin = out + (size_t)TT * BB * HH;            // [B,H]
    float* cfin = hfin + (size_t)BB * HH;                // [B,H]

    char* ws = (char*)d_ws;
    _Float16* WihH  = (_Float16*)(ws);                    // 512 KB
    _Float16* WhhH  = (_Float16*)(ws + (size_t)524288);   // 512 KB
    _Float16* WaH   = (_Float16*)(ws + (size_t)1048576);  // 256 KB
    float*    pre   = (float*)(ws + (size_t)1310720);     // 64 MB (tile-major)
    _Float16* hidsH = (_Float16*)(ws + (size_t)68419584); // 8 MB f16 history

    f32_to_f16_kernel<<<(GG * II + 255) / 256, 256, 0, stream>>>(Wih, WihH, GG * II);
    f32_to_f16_kernel<<<(GG * HH + 255) / 256, 256, 0, stream>>>(Whh, WhhH, GG * HH);
    f32_to_f16_kernel<<<(HH * 2 * HH + 255) / 256, 256, 0, stream>>>(Wa, WaH, HH * 2 * HH);

    // 65536 tiles (1024 M-tiles x 64 N-tiles), 4 waves per 128-thread block
    pre_gates_kernel<<<16384, 128, 0, stream>>>(embs, WihH, bih, bhh, pre);

    // 8 independent batch-tile recurrences
    alstm_kernel<<<BB / BT, NTHR, 0, stream>>>(lens, WhhH, WaH, ba, pre,
                                               hidsH, hs, hfin, cfin);
}
